// Generator_49435073577345
// MI455X (gfx1250) — compile-verified, matrix operand order checked by reading, not run on hardware
//
#include <hip/hip_runtime.h>
#include <hip/hip_bf16.h>
#include <stdint.h>

// ---------------- problem constants ----------------
#define B_     128
#define T_     16
#define H_     256
#define NOISE_ 128
#define NTOT   50000
#define C_     4
#define EMB_   128
#define TAU_   3.0f

typedef __bf16 bf16_t;
typedef __attribute__((ext_vector_type(16))) __bf16 v16bf;
typedef __attribute__((ext_vector_type(8)))  float  v8f;

// ---------------- device helpers ----------------
__device__ __forceinline__ v8f wmma_bf16(v16bf a, v16bf b, v8f c) {
  // D = A(16x32) * B(32x16) + C(16x16 f32)
  return __builtin_amdgcn_wmma_f32_16x16x32_bf16(
      false, a, false, b, (short)0, c, false, false);
}

// Load one 16x32 (A) or 32x16 (B) bf16 fragment.
// Per documented CDNA5 layout, each lane holds 16 bf16 as two contiguous
// 8-element runs: k = k0+hb+[0..7] and k = k0+16+hb+[0..7], hb = (lane&16)?8:0.
// Caller supplies row = tileBase + (lane & 15).
__device__ __forceinline__ v16bf load_frag(const bf16_t* __restrict__ base,
                                           int row, int stride, int k0) {
  int lane = threadIdx.x & 31;
  int hb = (lane & 16) ? 8 : 0;
  const bf16_t* p = base + (size_t)row * stride + k0 + hb;
  v16bf v;
#pragma unroll
  for (int e = 0; e < 8; ++e) { v[e] = p[e]; v[8 + e] = p[16 + e]; }
  return v;
}

__device__ __forceinline__ float sigf(float x) { return 1.0f / (1.0f + __expf(-x)); }

__device__ __forceinline__ unsigned hash3(unsigned a, unsigned b, unsigned c) {
  unsigned h = a * 0x9E3779B1u ^ (b + 0x7F4A7C15u) * 0x85EBCA77u
                               ^ (c + 0x165667B1u) * 0xC2B2AE3Du;
  h ^= h >> 16; h *= 0x7FEB352Du; h ^= h >> 15; h *= 0x846CA68Bu; h ^= h >> 16;
  return h;
}

__device__ __forceinline__ float gumbelf(unsigned a, unsigned b, unsigned c) {
  unsigned s = hash3(a, b, c);
  float u = ((s >> 9) + 0.5f) * (1.0f / 8388608.0f);  // (0,1)
  return -__logf(-__logf(u));
}

__device__ __forceinline__ unsigned orderable(float f) {
  unsigned u = __float_as_uint(f);
  return (u & 0x80000000u) ? ~u : (u | 0x80000000u);
}

__device__ __forceinline__ unsigned long long shfl_xor_u64(unsigned long long v,
                                                           int mask) {
  unsigned lo = (unsigned)v, hi = (unsigned)(v >> 32);
  lo = __shfl_xor(lo, mask, 16);
  hi = __shfl_xor(hi, mask, 16);
  return ((unsigned long long)hi << 32) | lo;
}

// Raw LDS byte offset of a pointer into a __shared__ object (addrspace(3)
// ptrtoint yields the wave-relative LDS offset the async VDST operand wants).
__device__ __forceinline__ unsigned lds_offset(const void* p) {
  return (unsigned)(size_t)(__attribute__((address_space(3))) const char*)p;
}

// CDNA5 async copy: 16 bytes per lane, global -> LDS, tracked by ASYNCcnt.
__device__ __forceinline__ void async_copy_b128(unsigned lds_addr,
                                                const void* gaddr) {
  asm volatile("global_load_async_to_lds_b128 %0, %1, off"
               :: "v"(lds_addr), "v"((unsigned long long)gaddr)
               : "memory");
}

__device__ __forceinline__ void wait_asynccnt0() {
  asm volatile("s_wait_asynccnt 0x0" ::: "memory");
}

// ---------------- utility kernels ----------------
__global__ void zero_kernel(float* p, long long n) {
  long long i = (long long)blockIdx.x * blockDim.x + threadIdx.x;
  long long n4 = n >> 2;
  if (i < n4) ((float4*)p)[i] = make_float4(0.f, 0.f, 0.f, 0.f);
  if (i == 0) for (long long k = n4 << 2; k < n; ++k) p[k] = 0.f;
}

__global__ void cvt_kernel(const float* __restrict__ s, bf16_t* __restrict__ d, int n) {
  int i = blockIdx.x * blockDim.x + threadIdx.x;
  if (i < n) d[i] = (bf16_t)s[i];
}

__global__ void biascomb_kernel(const float* a, const float* b, float* d, int n) {
  int i = blockIdx.x * blockDim.x + threadIdx.x;
  if (i < n) d[i] = a[i] + b[i];
}

// convert node-embedding table to bf16 and compute row norms; pad rows -> +huge
__global__ void embnorm_kernel(const float* __restrict__ src, bf16_t* __restrict__ dst,
                               float* __restrict__ norm, int rows, int padrows) {
  int r = blockIdx.x * blockDim.x + threadIdx.x;
  if (r >= padrows) return;
  if (r < rows) {
    float s = 0.f;
#pragma unroll 4
    for (int e = 0; e < EMB_; ++e) {
      float f = src[(size_t)r * EMB_ + e];
      dst[(size_t)r * EMB_ + e] = (bf16_t)f;
      s += f * f;
    }
    norm[r] = s;
  } else {
    for (int e = 0; e < EMB_; ++e) dst[(size_t)r * EMB_ + e] = (bf16_t)0.f;
    norm[r] = 3.0e38f;
  }
}

// inter = tanh(z @ W1^T + b1)
__global__ void init1_kernel(const float* __restrict__ z, const float* __restrict__ W,
                             const float* __restrict__ b, float* __restrict__ inter) {
  int tid = blockIdx.x * blockDim.x + threadIdx.x;
  if (tid >= B_ * H_) return;
  int bb = tid >> 8, j = tid & 255;
  float s = b[j];
#pragma unroll 4
  for (int k = 0; k < NOISE_; ++k) s += z[bb * NOISE_ + k] * W[j * NOISE_ + k];
  inter[tid] = tanhf(s);
}

// h0 = tanh(lin_2_c(inter)); c0 = tanh(lin_2_h(inter))   (reference swaps them)
__global__ void init2_kernel(const float* __restrict__ inter,
                             const float* __restrict__ Wc, const float* __restrict__ bc,
                             const float* __restrict__ Wh, const float* __restrict__ bh,
                             float* __restrict__ h, float* __restrict__ c,
                             bf16_t* __restrict__ hbf,
                             float* __restrict__ inp, bf16_t* __restrict__ inpbf) {
  int tid = blockIdx.x * blockDim.x + threadIdx.x;
  if (tid >= B_ * H_) return;
  int bb = tid >> 8, j = tid & 255;
  float sh = bc[j], sc = bh[j];
#pragma unroll 4
  for (int k = 0; k < H_; ++k) {
    float f = inter[bb * H_ + k];
    sh += f * Wc[j * H_ + k];
    sc += f * Wh[j * H_ + k];
  }
  float h0 = tanhf(sh), c0 = tanhf(sc);
  h[tid] = h0; h[B_ * H_ + tid] = h0;
  c[tid] = c0; c[B_ * H_ + tid] = c0;
  hbf[tid] = (bf16_t)h0; hbf[B_ * H_ + tid] = (bf16_t)h0;
  inp[tid] = 0.f; inpbf[tid] = (bf16_t)0.f;
}

// ---------------- fused LSTM layer step (WMMA) ----------------
// gates = x@Wih^T + h@Whh^T + (bih+bhh); then elementwise LSTM update.
// grid = (8 m-tiles, 16 j-groups), block = 32 (one wave owns all four gate
// columns for its 16x16 output footprint so the update stays in-register).
__global__ __launch_bounds__(32)
void lstm_kernel(const bf16_t* __restrict__ xbf, const bf16_t* __restrict__ hprevbf,
                 const float* __restrict__ cprev,
                 const bf16_t* __restrict__ wih, const bf16_t* __restrict__ whh,
                 const float* __restrict__ bias,
                 float* __restrict__ hnext, bf16_t* __restrict__ hnextbf,
                 float* __restrict__ cnext) {
  int lane = threadIdx.x;
  int m0 = blockIdx.x * 16;
  int jb = blockIdx.y * 16;
  int arow = m0 + (lane & 15);
  int brow = jb + (lane & 15);

  v8f acc[4];
#pragma unroll
  for (int g = 0; g < 4; ++g) acc[g] = (v8f){};

#pragma unroll
  for (int kc = 0; kc < 8; ++kc) {           // x @ Wih^T, K=256
    v16bf a = load_frag(xbf, arow, H_, kc * 32);
#pragma unroll
    for (int g = 0; g < 4; ++g) {
      v16bf b = load_frag(wih, g * H_ + brow, H_, kc * 32);
      acc[g] = wmma_bf16(a, b, acc[g]);
    }
  }
#pragma unroll
  for (int kc = 0; kc < 8; ++kc) {           // h @ Whh^T, K=256
    v16bf a = load_frag(hprevbf, arow, H_, kc * 32);
#pragma unroll
    for (int g = 0; g < 4; ++g) {
      v16bf b = load_frag(whh, g * H_ + brow, H_, kc * 32);
      acc[g] = wmma_bf16(a, b, acc[g]);
    }
  }

  int half = (lane & 16) ? 8 : 0;
  int j = jb + (lane & 15);
#pragma unroll
  for (int r = 0; r < 8; ++r) {
    int m = m0 + r + half;
    float ig = acc[0][r] + bias[0 * H_ + j];
    float fg = acc[1][r] + bias[1 * H_ + j];
    float gg = acc[2][r] + bias[2 * H_ + j];
    float og = acc[3][r] + bias[3 * H_ + j];
    float cp = cprev[m * H_ + j];
    float cn = sigf(fg) * cp + sigf(ig) * tanhf(gg);
    float hn = sigf(og) * tanhf(cn);
    cnext[m * H_ + j] = cn;
    hnext[m * H_ + j] = hn;
    hnextbf[m * H_ + j] = (bf16_t)hn;
  }
}

// ---------------- node projections (WMMA): proj_t = out @ Wnode_t^T + b ----------------
// grid = (8 m-tiles, 8 n-tiles, 3 tables), block = 32
__global__ __launch_bounds__(32)
void proj_kernel(const bf16_t* __restrict__ h1bf, const bf16_t* __restrict__ nodeW,
                 const float* __restrict__ b0, const float* __restrict__ b1,
                 const float* __restrict__ b2,
                 float* __restrict__ proj, bf16_t* __restrict__ projbf) {
  int lane = threadIdx.x;
  int t = blockIdx.z, m0 = blockIdx.x * 16, n0 = blockIdx.y * 16;
  const bf16_t* W = nodeW + (size_t)t * EMB_ * H_;
  v8f acc = (v8f){};
#pragma unroll
  for (int kc = 0; kc < 8; ++kc) {
    v16bf a = load_frag(h1bf, m0 + (lane & 15), H_, kc * 32);
    v16bf b = load_frag(W, n0 + (lane & 15), H_, kc * 32);
    acc = wmma_bf16(a, b, acc);
  }
  const float* bb = (t == 0) ? b0 : ((t == 1) ? b1 : b2);
  int n = n0 + (lane & 15);
  int half = (lane & 16) ? 8 : 0;
#pragma unroll
  for (int r = 0; r < 8; ++r) {
    int m = m0 + r + half;
    float v = acc[r] + bb[n];
    proj[((size_t)t * B_ + m) * EMB_ + n] = v;
    projbf[((size_t)t * B_ + m) * EMB_ + n] = (bf16_t)v;
  }
}

// ---------------- type head: logits + Gumbel argmax; proj norms; reset argmax ----------------
__global__ void type_kernel(const float* __restrict__ h1, const float* __restrict__ Wt,
                            const float* __restrict__ bt, const float* __restrict__ proj,
                            float* __restrict__ pnorm, int* __restrict__ hard,
                            unsigned long long* __restrict__ amax,
                            float* __restrict__ dout, int step) {
  int b = threadIdx.x;  // 128 threads
  amax[b] = 0ull; amax[128 + b] = 0ull; amax[256 + b] = 0ull;

  float best = -3.4e38f; int bi = 0;
#pragma unroll
  for (int c = 0; c < C_; ++c) {
    float s = bt[c];
#pragma unroll 4
    for (int k = 0; k < H_; ++k) s += h1[b * H_ + k] * Wt[c * H_ + k];
    float v = s + gumbelf(0x51E9u + (unsigned)step, (unsigned)b, (unsigned)c);
    if (v > best) { best = v; bi = c; }
  }
  hard[b] = bi;
  // output_type == y_hard exactly (soft terms cancel in value)
  dout[(size_t)b * T_ * C_ + (size_t)step * C_ + bi] = 1.0f;

#pragma unroll
  for (int t3 = 0; t3 < 3; ++t3) {
    float s = 0.f;
#pragma unroll 4
    for (int e = 0; e < EMB_; ++e) {
      float p = proj[((size_t)t3 * B_ + b) * EMB_ + e];
      s += p * p;
    }
    pnorm[t3 * B_ + b] = s;
  }
}

// ---------------- distance GEMM + Gumbel-max argmax (WMMA + async LDS) ----------------
// grid = padded n-tiles of this table; block = 128 (4 waves x 2 m-tiles each).
// The 16 embedding rows (4 KB) for this n-tile are staged into LDS once per
// block with GLOBAL_LOAD_ASYNC_TO_LDS_B128 (ASYNCcnt) and shared by all 4
// waves' B fragments, instead of 4x redundant global fetches.
__global__ __launch_bounds__(128)
void dist_kernel(const bf16_t* __restrict__ projbf_t, const float* __restrict__ pnorm_t,
                 const bf16_t* __restrict__ emb, const float* __restrict__ tn,
                 int nreal, unsigned long long* __restrict__ amax_t,
                 int table, int step) {
  __shared__ __align__(16) bf16_t tileB[16 * EMB_];  // 16 rows x 128 bf16 = 4 KB

  int tid = threadIdx.x;
  int wave = tid >> 5, lane = tid & 31;
  int n0 = blockIdx.x * 16;
  int nrow = n0 + (lane & 15);

  // stage B tile: 256 x 16B chunks, 2 per thread, async global->LDS
  {
    const char* gbase = (const char*)(emb + (size_t)n0 * EMB_);
    unsigned lbase = lds_offset(tileB);
#pragma unroll
    for (int c = 0; c < 2; ++c) {
      int chunk = tid + c * 128;  // 0..255
      async_copy_b128(lbase + chunk * 16, gbase + chunk * 16);
    }
  }
  if (blockIdx.x + 1 < gridDim.x)  // prefetch next tile's embedding rows (L2)
    __builtin_prefetch(emb + (size_t)(nrow + 16) * EMB_, 0, 1);
  wait_asynccnt0();   // our wave's async copies landed in LDS
  __syncthreads();    // all waves' portions visible

  int m0a = (2 * wave) * 16, m0b = m0a + 16;
  v8f acc0 = (v8f){}, acc1 = (v8f){};
#pragma unroll
  for (int kc = 0; kc < 4; ++kc) {          // K = EMB = 128
    v16bf bf = load_frag(tileB, lane & 15, EMB_, kc * 32);  // ds_load path
    v16bf a0 = load_frag(projbf_t, m0a + (lane & 15), EMB_, kc * 32);
    v16bf a1 = load_frag(projbf_t, m0b + (lane & 15), EMB_, kc * 32);
    acc0 = wmma_bf16(a0, bf, acc0);
    acc1 = wmma_bf16(a1, bf, acc1);
  }

  float tnv = tn[nrow];
  int oob = (nrow >= nreal);
  int half = (lane & 16) ? 8 : 0;
#pragma unroll
  for (int pair = 0; pair < 2; ++pair) {
    v8f acc = pair ? acc1 : acc0;
    int m0 = pair ? m0b : m0a;
#pragma unroll
    for (int r = 0; r < 8; ++r) {
      int m = m0 + r + half;
      float sq = pnorm_t[m] - 2.0f * acc[r] + tnv;
      float d = sqrtf(fmaxf(sq, 0.0f));
      float sc = gumbelf(0xABCDu + (unsigned)table * 0x1000u + (unsigned)step,
                         (unsigned)m, (unsigned)nrow) - d;
      if (oob) sc = -3.0e38f;
      unsigned long long key =
          ((unsigned long long)orderable(sc) << 32) | (unsigned)nrow;
      // reduce over the 16 lanes that share batch row m
#pragma unroll
      for (int off = 1; off < 16; off <<= 1) {
        unsigned long long o = shfl_xor_u64(key, off);
        if (o > key) key = o;
      }
      if ((lane & 15) == 0) atomicMax(amax_t + m, key);
    }
  }
}

// ---------------- select node, scatter one-hot, build inp_next ----------------
// grid = 128 (batch), block = 256 (hidden)
__global__ __launch_bounds__(256)
void select_kernel(const unsigned long long* __restrict__ amax,
                   const int* __restrict__ hard,
                   const int* __restrict__ tt0, const int* __restrict__ tt1,
                   const int* __restrict__ tt2,
                   const float* __restrict__ Wdt, const float* __restrict__ bdt,
                   const float* __restrict__ Wdn, const float* __restrict__ bdn,
                   float* __restrict__ inp, bf16_t* __restrict__ inpbf,
                   float* __restrict__ dout, int step) {
  int b = blockIdx.x, i = threadIdx.x;
  int hi = hard[b];
  int node;
  if (hi == 0)      node = tt0[(int)(amax[0 * B_ + b] & 0xffffffffull)];
  else if (hi == 1) node = tt1[(int)(amax[1 * B_ + b] & 0xffffffffull)];
  else if (hi == 2) node = tt2[(int)(amax[2 * B_ + b] & 0xffffffffull)];
  else              node = NTOT;

  float v = Wdt[i * C_ + hi] + bdt[i] + Wdn[(size_t)i * (NTOT + 1) + node] + bdn[i];
  inp[b * H_ + i] = v;
  inpbf[b * H_ + i] = (bf16_t)v;

  if (i == 0) {
    float* outNode = dout + (size_t)B_ * T_ * C_;
    outNode[(size_t)b * T_ * (NTOT + 1) + (size_t)step * (NTOT + 1) + node] = 1.0f;
    float* outIdx = dout + (size_t)B_ * T_ * C_ + (size_t)B_ * T_ * (NTOT + 1);
    outIdx[step * B_ + b] = (float)node;
  }
}

// ---------------- host side ----------------
enum {
  IN_Z = 0, IN_L1W, IN_L1B, IN_L2HW, IN_L2HB, IN_L2CW, IN_L2CB,
  IN_TYPEW, IN_TYPEB,
  IN_N0W, IN_N0B, IN_N1W, IN_N1B, IN_N2W, IN_N2B,
  IN_WDTW, IN_WDTB, IN_WDNW, IN_WDNB,
  IN_L0_WIH, IN_L0_WHH, IN_L0_BIH, IN_L0_BHH,
  IN_L1_WIH, IN_L1_WHH, IN_L1_BIH, IN_L1_BHH,
  IN_EMB0, IN_EMB1, IN_EMB2, IN_TT0, IN_TT1, IN_TT2
};

extern "C" void kernel_launch(void* const* d_in, const int* in_sizes, int n_in,
                              void* d_out, int out_size, void* d_ws, size_t ws_size,
                              hipStream_t stream) {
  (void)in_sizes; (void)n_in; (void)ws_size;
  const float* F[33];
  for (int i = 0; i < 33; ++i) F[i] = (const float*)d_in[i];

  // ---- workspace carve-up ----
  uintptr_t cur = (uintptr_t)d_ws;
  auto alloc = [&](size_t bytes) -> void* {
    void* p = (void*)cur;
    cur += (bytes + 255) & ~(size_t)255;
    return p;
  };
  const int rows[3]    = {17000, 17000, 16000};
  const int padRows[3] = {17008, 17008, 16000};
  const int rowOff[3]  = {0, 17008, 34016};
  const int nTiles[3]  = {1063, 1063, 1000};
  const size_t PADTOT = 50016;

  bf16_t* lstmWbf = (bf16_t*)alloc(4 * 1024 * 256 * sizeof(bf16_t)); // [l][Wih|Whh]
  float*  biasL   = (float*)alloc(2 * 1024 * sizeof(float));
  bf16_t* nodeWbf = (bf16_t*)alloc(3 * EMB_ * H_ * sizeof(bf16_t));
  bf16_t* embbf   = (bf16_t*)alloc(PADTOT * EMB_ * sizeof(bf16_t));
  float*  tnorm   = (float*)alloc(PADTOT * sizeof(float));
  float*  hA = (float*)alloc(2 * B_ * H_ * sizeof(float));
  float*  hB = (float*)alloc(2 * B_ * H_ * sizeof(float));
  float*  cA = (float*)alloc(2 * B_ * H_ * sizeof(float));
  float*  cB = (float*)alloc(2 * B_ * H_ * sizeof(float));
  bf16_t* hbfA = (bf16_t*)alloc(2 * B_ * H_ * sizeof(bf16_t));
  bf16_t* hbfB = (bf16_t*)alloc(2 * B_ * H_ * sizeof(bf16_t));
  float*  inter = (float*)alloc(B_ * H_ * sizeof(float));
  float*  inp   = (float*)alloc(B_ * H_ * sizeof(float));
  bf16_t* inpbf = (bf16_t*)alloc(B_ * H_ * sizeof(bf16_t));
  float*  proj  = (float*)alloc(3 * B_ * EMB_ * sizeof(float));
  bf16_t* projbf = (bf16_t*)alloc(3 * B_ * EMB_ * sizeof(bf16_t));
  float*  pnorm = (float*)alloc(3 * B_ * sizeof(float));
  int*    hard  = (int*)alloc(B_ * sizeof(int));
  unsigned long long* amax = (unsigned long long*)alloc(3 * B_ * sizeof(unsigned long long));

  float* dout = (float*)d_out;

  // ---- zero full output (one-hots are scattered later) ----
  {
    long long n = (long long)out_size;
    long long n4 = n >> 2;
    int blocks = (int)((n4 + 255) / 256);
    zero_kernel<<<blocks, 256, 0, stream>>>(dout, n);
  }

  // ---- weight conversion / precompute (every call; deterministic) ----
  {
    const int nW = 1024 * 256;
    cvt_kernel<<<(nW + 255) / 256, 256, 0, stream>>>(F[IN_L0_WIH], lstmWbf + 0 * nW, nW);
    cvt_kernel<<<(nW + 255) / 256, 256, 0, stream>>>(F[IN_L0_WHH], lstmWbf + 1 * nW, nW);
    cvt_kernel<<<(nW + 255) / 256, 256, 0, stream>>>(F[IN_L1_WIH], lstmWbf + 2 * nW, nW);
    cvt_kernel<<<(nW + 255) / 256, 256, 0, stream>>>(F[IN_L1_WHH], lstmWbf + 3 * nW, nW);
    biascomb_kernel<<<4, 256, 0, stream>>>(F[IN_L0_BIH], F[IN_L0_BHH], biasL, 1024);
    biascomb_kernel<<<4, 256, 0, stream>>>(F[IN_L1_BIH], F[IN_L1_BHH], biasL + 1024, 1024);
    const int nN = EMB_ * H_;
    cvt_kernel<<<(nN + 255) / 256, 256, 0, stream>>>(F[IN_N0W], nodeWbf + 0 * nN, nN);
    cvt_kernel<<<(nN + 255) / 256, 256, 0, stream>>>(F[IN_N1W], nodeWbf + 1 * nN, nN);
    cvt_kernel<<<(nN + 255) / 256, 256, 0, stream>>>(F[IN_N2W], nodeWbf + 2 * nN, nN);
    const float* embsrc[3] = {F[IN_EMB0], F[IN_EMB1], F[IN_EMB2]};
    for (int t = 0; t < 3; ++t)
      embnorm_kernel<<<(padRows[t] + 127) / 128, 128, 0, stream>>>(
          embsrc[t], embbf + (size_t)rowOff[t] * EMB_, tnorm + rowOff[t],
          rows[t], padRows[t]);
  }

  // ---- initial state ----
  init1_kernel<<<(B_ * H_ + 255) / 256, 256, 0, stream>>>(F[IN_Z], F[IN_L1W], F[IN_L1B], inter);
  init2_kernel<<<(B_ * H_ + 255) / 256, 256, 0, stream>>>(
      inter, F[IN_L2CW], F[IN_L2CB], F[IN_L2HW], F[IN_L2HB],
      hA, cA, hbfA, inp, inpbf);

  // ---- 16 autoregressive steps ----
  for (int t = 0; t < T_; ++t) {
    float*  hp  = (t & 1) ? hB : hA;    float*  hn  = (t & 1) ? hA : hB;
    float*  cp  = (t & 1) ? cB : cA;    float*  cn  = (t & 1) ? cA : cB;
    bf16_t* hbp = (t & 1) ? hbfB : hbfA; bf16_t* hbn = (t & 1) ? hbfA : hbfB;
    (void)hp;
    const int nW = 1024 * 256;

    // layer 0: x = inp
    lstm_kernel<<<dim3(8, 16), 32, 0, stream>>>(
        inpbf, hbp + 0 * B_ * H_, cp + 0 * B_ * H_,
        lstmWbf + 0 * nW, lstmWbf + 1 * nW, biasL + 0,
        hn + 0 * B_ * H_, hbn + 0 * B_ * H_, cn + 0 * B_ * H_);
    // layer 1: x = new h of layer 0
    lstm_kernel<<<dim3(8, 16), 32, 0, stream>>>(
        hbn + 0 * B_ * H_, hbp + 1 * B_ * H_, cp + 1 * B_ * H_,
        lstmWbf + 2 * nW, lstmWbf + 3 * nW, biasL + 1024,
        hn + 1 * B_ * H_, hbn + 1 * B_ * H_, cn + 1 * B_ * H_);

    proj_kernel<<<dim3(8, 8, 3), 32, 0, stream>>>(
        hbn + 1 * B_ * H_, nodeWbf, F[IN_N0B], F[IN_N1B], F[IN_N2B], proj, projbf);

    type_kernel<<<1, 128, 0, stream>>>(
        hn + 1 * B_ * H_, F[IN_TYPEW], F[IN_TYPEB], proj, pnorm, hard, amax, dout, t);

    for (int tb = 0; tb < 3; ++tb)
      dist_kernel<<<nTiles[tb], 128, 0, stream>>>(
          projbf + (size_t)tb * B_ * EMB_, pnorm + tb * B_,
          embbf + (size_t)rowOff[tb] * EMB_, tnorm + rowOff[tb],
          rows[tb], amax + tb * B_, tb, t);

    select_kernel<<<B_, H_, 0, stream>>>(
        amax, hard, (const int*)d_in[IN_TT0], (const int*)d_in[IN_TT1],
        (const int*)d_in[IN_TT2], F[IN_WDTW], F[IN_WDTB], F[IN_WDNW], F[IN_WDNB],
        inp, inpbf, dout, t);
  }
}
